// Mixup_Branch_29102698397869
// MI455X (gfx1250) — matrix operand order repaired and checked by reading, not compiled, not run on hardware
//
#include <hip/hip_runtime.h>

// ---------------------------------------------------------------------------
// Problem constants (match reference setup_inputs)
// ---------------------------------------------------------------------------
#define BATCH 4
#define CIN   512     // C
#define PDIM  512     // P
#define TDIM  2048    // T
#define TFDIM 4096    // TF
#define M12   1536    // P + 2P rows of fused GEMM1+2
#define CMIX  2048    // channels of "mixed" (sampled 512 | feat 1024 | fm 512)
#define NGROUPS 32
#define GN_EPS 1e-5f

typedef __attribute__((ext_vector_type(16))) __bf16 v16bf;
typedef __attribute__((ext_vector_type(8)))  __bf16 v8bf;
typedef __attribute__((ext_vector_type(8)))  float  v8f;
typedef __attribute__((ext_vector_type(4)))  int    v4i;

// ---------------------------------------------------------------------------
// gfx1250 async global->LDS path (ASYNCcnt), with portable fallback
// ---------------------------------------------------------------------------
#if defined(__has_builtin)
#  if __has_builtin(__builtin_amdgcn_global_load_async_to_lds_b128)
#    define HAVE_ASYNC_LDS 1
#  endif
#endif
#ifndef HAVE_ASYNC_LDS
#  define HAVE_ASYNC_LDS 0
#endif

__device__ __forceinline__ void wait_async_zero() {
#if HAVE_ASYNC_LDS
#  if defined(__has_builtin) && __has_builtin(__builtin_amdgcn_s_wait_asynccnt)
    __builtin_amdgcn_s_wait_asynccnt(0);
#  else
    asm volatile("s_wait_asynccnt 0x0" ::: "memory");
#  endif
#endif
}

// Stage one 32-byte (16 x bf16) chunk from global into LDS.
// Builtin signature (probe-confirmed via round-2 diagnostic):
//   void (v4i __attribute__((address_space(1)))* src,
//         v4i __attribute__((address_space(3)))* dst, imm offset, imm cpol)
__device__ __forceinline__ void stage_b_chunk(const __bf16* __restrict__ gsrc,
                                              __bf16* lds_dst) {
#if HAVE_ASYNC_LDS
    __builtin_amdgcn_global_load_async_to_lds_b128(
        (__attribute__((address_space(1))) v4i*)gsrc,
        (__attribute__((address_space(3))) v4i*)lds_dst, 0, 0);
    __builtin_amdgcn_global_load_async_to_lds_b128(
        (__attribute__((address_space(1))) v4i*)(gsrc + 8),
        (__attribute__((address_space(3))) v4i*)(lds_dst + 8), 0, 0);
#else
    *(v8bf*)lds_dst       = *(const v8bf*)gsrc;
    *(v8bf*)(lds_dst + 8) = *(const v8bf*)(gsrc + 8);
#endif
}

// ---------------------------------------------------------------------------
// WMMA fragment loads (layouts per CDNA5 ISA 7.12.2, wave32)
// A (16x32 bf16, row-major src, ld = K):
//   lane r in [0,16): M = r; elems 0..7 = K k0+h*8.. ; elems 8..15 = K k0+16+h*8..
// B (32x16 bf16): lane: N = r; elems 0..7 = K h*8.. ; elems 8..15 = K 16+h*8..
// ---------------------------------------------------------------------------
__device__ __forceinline__ v16bf load_frag_a(const __bf16* __restrict__ A, int lda,
                                             int m0, int k0, int lane) {
    int r = lane & 15, h = lane >> 4;
    const __bf16* p = A + (size_t)(m0 + r) * lda + (k0 + h * 8);
    v8bf lo = *(const v8bf*)p;
    v8bf hi = *(const v8bf*)(p + 16);
    return __builtin_shufflevector(lo, hi, 0, 1, 2, 3, 4, 5, 6, 7,
                                           8, 9, 10, 11, 12, 13, 14, 15);
}

#define LDSROW 40   // bf16 elements per LDS B-tile row: 32 data + 8 pad (bank-safe)

__device__ __forceinline__ v16bf lds_frag_b(const __bf16* buf, int nloc, int lane) {
    int r = lane & 15, h = lane >> 4;
    const __bf16* p = buf + (nloc + r) * LDSROW + h * 8;
    v8bf lo = *(const v8bf*)p;
    v8bf hi = *(const v8bf*)(p + 16);
    return __builtin_shufflevector(lo, hi, 0, 1, 2, 3, 4, 5, 6, 7,
                                           8, 9, 10, 11, 12, 13, 14, 15);
}

// ---------------------------------------------------------------------------
// GEMM: Out[b, m, n] = sum_k A[m,k] * BT[b, n, k] + bias[m]
// A: (M x K) bf16 row-major (batch-shared weights, direct from global/L2)
// BT: (B, N, K) bf16 (activation; B tile double-buffered in LDS via async copy)
// Out: (B, M, N) f32
// Workgroup: 256 threads = 8 waves; wave tile 32x64; WG tile 128x128.
// ---------------------------------------------------------------------------
#define WTM 2   // 16x16 frags along M per wave
#define WTN 4   // 16x16 frags along N per wave

__global__ void __launch_bounds__(256)
gemm_bf16_wmma(const __bf16* __restrict__ A,
               const __bf16* __restrict__ BT,
               const float* __restrict__ bias,
               float* __restrict__ Out,
               int M, int N, int K,
               size_t strideB_bt, size_t strideB_out) {
    __shared__ __bf16 bbuf[2][128 * LDSROW];   // 2 x 10 KB

    const int b    = blockIdx.z;
    const int tid  = threadIdx.x;
    const int wave = tid >> 5;
    const int lane = tid & 31;
    const int wm   = wave & 3;                 // 4 waves along M
    const int wn   = wave >> 2;                // 2 waves along N
    const int m_base  = blockIdx.y * 128 + wm * 32;
    const int n_local = wn * 64;               // wave's N offset inside WG tile

    const __bf16* Bp = BT + (size_t)b * strideB_bt;
    float*        Op = Out + (size_t)b * strideB_out;

    // per-thread B-tile staging assignment: 128 rows x two 32B half-rows
    const int crow  = tid >> 1;                // 0..127  (local n)
    const int chalf = (tid & 1) * 16;          // element offset within row
    const __bf16* gB_row = Bp + (size_t)(blockIdx.x * 128 + crow) * K + chalf;
    __bf16* ldsd0 = &bbuf[0][crow * LDSROW + chalf];
    __bf16* ldsd1 = &bbuf[1][crow * LDSROW + chalf];

    v8f acc[WTM][WTN] = {};

    // prologue: stage first k-tile into buffer 0
    stage_b_chunk(gB_row, ldsd0);

    int cur = 0;
    for (int k0 = 0; k0 < K; k0 += 32) {
        wait_async_zero();       // this wave's staging transfers complete
        __syncthreads();         // tile visible to all waves

        const int knext = k0 + 32;
        if (knext < K)
            stage_b_chunk(gB_row + knext, cur ? ldsd0 : ldsd1);

        const __bf16* btile = bbuf[cur];
        v16bf af[WTM], bf[WTN];
#pragma unroll
        for (int i = 0; i < WTM; ++i)
            af[i] = load_frag_a(A, K, m_base + 16 * i, k0, lane);
#pragma unroll
        for (int j = 0; j < WTN; ++j)
            bf[j] = lds_frag_b(btile, n_local + 16 * j, lane);
#pragma unroll
        for (int i = 0; i < WTM; ++i)
#pragma unroll
            for (int j = 0; j < WTN; ++j)
                acc[i][j] = __builtin_amdgcn_wmma_f32_16x16x32_bf16(
                    false, af[i], false, bf[j], (short)0, acc[i][j], false, false);

        __syncthreads();         // all reads of bbuf[cur] done before reuse
        cur ^= 1;
    }

    // Epilogue: C/D layout -> elem e is row m_base+16i+8*h+e, col n_base+16j+r
    const int r = lane & 15, h = lane >> 4;
    const int n_base = blockIdx.x * 128 + n_local;
#pragma unroll
    for (int i = 0; i < WTM; ++i) {
        const int row0 = m_base + 16 * i + h * 8;
#pragma unroll
        for (int j = 0; j < WTN; ++j) {
            const int col = n_base + 16 * j + r;
#pragma unroll
            for (int e = 0; e < 8; ++e) {
                Op[(size_t)(row0 + e) * N + col] = acc[i][j][e] + bias[row0 + e];
            }
        }
    }
}

// ---------------------------------------------------------------------------
// Sampler: mean over channels (batch 0) -> normalized cumsum -> cdf ints ->
// idx[t] = first argmin_j |cdf[j] - t|.   Single workgroup.
// ---------------------------------------------------------------------------
__global__ void __launch_bounds__(256)
sampler_kernel(const float* __restrict__ flf, int* __restrict__ idx) {
    __shared__ float msum[TFDIM];
    __shared__ int   cdf[TFDIM];
    const int tid = threadIdx.x;

    for (int j = tid; j < TFDIM; j += 256) {
        float s = 0.f;
        for (int c = 0; c < CIN; ++c)
            s += flf[(size_t)c * TFDIM + j];
        msum[j] = s;                       // 1/C factor cancels in normalization
    }
    __syncthreads();

    if (tid == 0) {
        float tot = 0.f;
        for (int j = 0; j < TFDIM; ++j) tot += msum[j];
        const float inv = 1.f / tot;
        float run = 0.f;
        for (int j = 0; j < TFDIM; ++j) {
            run += msum[j];
            int v = (int)(run * inv * (float)TDIM);   // trunc, values >= 0
            cdf[j] = v < (TDIM - 1) ? v : (TDIM - 1);
        }
    }
    __syncthreads();

    for (int t = tid; t < TDIM; t += 256) {
        int best = 0;
        int bd = cdf[0] > t ? cdf[0] - t : t - cdf[0];
        for (int j = 1; j < TFDIM; ++j) {
            int d = cdf[j] > t ? cdf[j] - t : t - cdf[j];
            if (d < bd) { bd = d; best = j; }          // first occurrence wins
        }
        idx[t] = best;
    }
}

// ---------------------------------------------------------------------------
// Elementwise f32 -> bf16 convert (weights) and f32 copy (bias concat)
// ---------------------------------------------------------------------------
__global__ void f32_to_bf16_kernel(const float* __restrict__ in,
                                   __bf16* __restrict__ out, size_t n) {
    size_t i = (size_t)blockIdx.x * blockDim.x + threadIdx.x;
    size_t stride = (size_t)gridDim.x * blockDim.x;
    for (; i < n; i += stride) out[i] = (__bf16)in[i];
}

__global__ void copy_f32_kernel(const float* __restrict__ in,
                                float* __restrict__ out, int n) {
    int i = blockIdx.x * blockDim.x + threadIdx.x;
    if (i < n) out[i] = in[i];
}

// ---------------------------------------------------------------------------
// Transpose (B, C, T) f32 -> (B, T, C) bf16 via LDS 32x32 tile.
// block (32, 8); grid (T/32, C/32, B)
// ---------------------------------------------------------------------------
__global__ void transpose_to_bf16_kernel(const float* __restrict__ X,
                                         __bf16* __restrict__ XT) {
    __shared__ float tile[32][33];
    const int b  = blockIdx.z;
    const int t0 = blockIdx.x * 32, c0 = blockIdx.y * 32;
    const int tx = threadIdx.x, ty = threadIdx.y;
    const float* Xb  = X  + (size_t)b * CIN * TDIM;
    __bf16*      XTb = XT + (size_t)b * CIN * TDIM;
#pragma unroll
    for (int r = 0; r < 4; ++r) {
        const int c = c0 + ty + r * 8;
        tile[ty + r * 8][tx] = Xb[(size_t)c * TDIM + t0 + tx];
    }
    __syncthreads();
#pragma unroll
    for (int r = 0; r < 4; ++r) {
        const int t = t0 + ty + r * 8;
        XTb[(size_t)t * CIN + c0 + tx] = (__bf16)tile[tx][ty + r * 8];
    }
}

// ---------------------------------------------------------------------------
// Gather sampled = flf[:, :, idx] -> bf16 into mixedT[b, t, 0..511]
// block (32, 8); grid (T/32, C/32, B)
// ---------------------------------------------------------------------------
__global__ void gather_sampled_kernel(const float* __restrict__ flf,
                                      const int* __restrict__ idx,
                                      __bf16* __restrict__ mixedT) {
    __shared__ float tile[32][33];
    __shared__ int   id[32];
    const int b  = blockIdx.z;
    const int t0 = blockIdx.x * 32, c0 = blockIdx.y * 32;
    const int tx = threadIdx.x, ty = threadIdx.y;
    if (ty == 0) id[tx] = idx[t0 + tx];
    __syncthreads();
    const float* fb = flf + (size_t)b * CIN * TFDIM;
#pragma unroll
    for (int r = 0; r < 4; ++r) {
        const int c = c0 + ty + r * 8;
        tile[ty + r * 8][tx] = fb[(size_t)c * TFDIM + id[tx]];
    }
    __syncthreads();
    __bf16* mp = mixedT + (size_t)b * TDIM * CMIX;
#pragma unroll
    for (int r = 0; r < 4; ++r) {
        const int t = t0 + ty + r * 8;
        mp[(size_t)t * CMIX + c0 + tx] = (__bf16)tile[tx][ty + r * 8];
    }
}

// ---------------------------------------------------------------------------
// GroupNorm stats: one block per (group, batch). Y laid out (b, c, t) with
// per-batch stride ystrideB; group g covers chpg contiguous channels.
// stats[(b*32+g)*2] = mean, [...+1] = rsqrt(var + eps)
// ---------------------------------------------------------------------------
__global__ void __launch_bounds__(256)
gn_stats_kernel(const float* __restrict__ Y, size_t ystrideB,
                float* __restrict__ stats, int chpg) {
    const int b = blockIdx.z, g = blockIdx.x;
    const float* p = Y + (size_t)b * ystrideB + (size_t)g * chpg * TDIM;
    const size_t n = (size_t)chpg * TDIM;
    float s = 0.f, s2 = 0.f;
    for (size_t i = threadIdx.x; i < n; i += 256) {
        float v = p[i];
        s += v; s2 += v * v;
    }
    __shared__ float rs[256], rq[256];
    rs[threadIdx.x] = s; rq[threadIdx.x] = s2;
    __syncthreads();
    for (int off = 128; off > 0; off >>= 1) {
        if ((int)threadIdx.x < off) {
            rs[threadIdx.x] += rs[threadIdx.x + off];
            rq[threadIdx.x] += rq[threadIdx.x + off];
        }
        __syncthreads();
    }
    if (threadIdx.x == 0) {
        float invn = 1.f / (float)n;
        float mu  = rs[0] * invn;
        float var = rq[0] * invn - mu * mu;
        stats[((size_t)b * NGROUPS + g) * 2 + 0] = mu;
        stats[((size_t)b * NGROUPS + g) * 2 + 1] = rsqrtf(var + GN_EPS);
    }
}

// ---------------------------------------------------------------------------
// GroupNorm apply + ReLU, optional f32 output, plus bf16 transposed write into
// mixedT[b, t, mixed_off + c].  block (32,8); grid (T/32, Cch/32, B)
// ---------------------------------------------------------------------------
__global__ void gn_apply_tr_kernel(const float* __restrict__ Y, size_t ystrideB,
                                   const float* __restrict__ stats,
                                   const float* __restrict__ gamma,
                                   const float* __restrict__ beta,
                                   float* __restrict__ out_f32, int outCch,
                                   __bf16* __restrict__ mixedT, int mixed_off,
                                   int chpg) {
    __shared__ float tile[32][33];
    const int b  = blockIdx.z;
    const int t0 = blockIdx.x * 32, c0 = blockIdx.y * 32;
    const int tx = threadIdx.x, ty = threadIdx.y;
    const float* Yb = Y + (size_t)b * ystrideB;
#pragma unroll
    for (int r = 0; r < 4; ++r) {
        const int c = c0 + ty + r * 8;
        const int g = c / chpg;
        const float mu   = stats[((size_t)b * NGROUPS + g) * 2 + 0];
        const float rstd = stats[((size_t)b * NGROUPS + g) * 2 + 1];
        float v = Yb[(size_t)c * TDIM + t0 + tx];
        v = (v - mu) * rstd * gamma[c] + beta[c];
        v = v > 0.f ? v : 0.f;
        if (out_f32)
            out_f32[((size_t)b * outCch + c) * TDIM + t0 + tx] = v;
        tile[ty + r * 8][tx] = v;
    }
    __syncthreads();
    __bf16* mp = mixedT + (size_t)b * TDIM * CMIX;
#pragma unroll
    for (int r = 0; r < 4; ++r) {
        const int t = t0 + ty + r * 8;
        mp[(size_t)t * CMIX + mixed_off + c0 + tx] = (__bf16)tile[tx][ty + r * 8];
    }
}

// ---------------------------------------------------------------------------
// Final GroupNorm apply + ReLU (elementwise).  grid (T/256, Cch, B)
// ---------------------------------------------------------------------------
__global__ void gn_apply_kernel(const float* __restrict__ Y,
                                const float* __restrict__ stats,
                                const float* __restrict__ gamma,
                                const float* __restrict__ beta,
                                float* __restrict__ out, int Cch, int chpg) {
    const int b = blockIdx.z, c = blockIdx.y;
    const int t = blockIdx.x * 256 + threadIdx.x;
    const int g = c / chpg;
    const float mu   = stats[((size_t)b * NGROUPS + g) * 2 + 0];
    const float rstd = stats[((size_t)b * NGROUPS + g) * 2 + 1];
    float v = Y[((size_t)b * Cch + c) * TDIM + t];
    v = (v - mu) * rstd * gamma[c] + beta[c];
    out[((size_t)b * Cch + c) * TDIM + t] = v > 0.f ? v : 0.f;
}

// ---------------------------------------------------------------------------
// Host launcher
// ---------------------------------------------------------------------------
static inline size_t align256(size_t x) { return (x + 255) & ~(size_t)255; }

extern "C" void kernel_launch(void* const* d_in, const int* in_sizes, int n_in,
                              void* d_out, int out_size, void* d_ws, size_t ws_size,
                              hipStream_t stream) {
    const float* feature = (const float*)d_in[0];   // (B, C, T)
    const float* flf     = (const float*)d_in[1];   // (B, C, TF)
    const float* W1 = (const float*)d_in[2];        // (P, C)
    const float* b1 = (const float*)d_in[3];
    const float* g1 = (const float*)d_in[4];
    const float* be1 = (const float*)d_in[5];
    const float* W2 = (const float*)d_in[6];        // (2P, C)
    const float* b2 = (const float*)d_in[7];
    const float* g2 = (const float*)d_in[8];
    const float* be2 = (const float*)d_in[9];
    const float* W3 = (const float*)d_in[10];       // (C, 4P)
    const float* b3 = (const float*)d_in[11];
    const float* g3 = (const float*)d_in[12];
    const float* be3 = (const float*)d_in[13];

    float* out_mixed = (float*)d_out;                                   // (B, C, T)
    float* out_feat  = (float*)d_out + (size_t)BATCH * CIN * TDIM;      // (B, 2P, T)

    // ---- workspace carve-up ----
    char* w = (char*)d_ws;
    int*    idx     = (int*)w;      w += align256((size_t)TDIM * 4);
    float*  biascat = (float*)w;    w += align256((size_t)M12 * 4);
    float*  stats1  = (float*)w;    w += align256((size_t)BATCH * NGROUPS * 2 * 4);
    float*  stats2  = (float*)w;    w += align256((size_t)BATCH * NGROUPS * 2 * 4);
    float*  stats3  = (float*)w;    w += align256((size_t)BATCH * NGROUPS * 2 * 4);
    __bf16* featT   = (__bf16*)w;   w += align256((size_t)BATCH * TDIM * CIN * 2);
    __bf16* Wcat    = (__bf16*)w;   w += align256((size_t)M12 * CIN * 2);
    __bf16* W3b     = (__bf16*)w;   w += align256((size_t)CIN * CMIX * 2);
    __bf16* mixedT  = (__bf16*)w;   w += align256((size_t)BATCH * TDIM * CMIX * 2);
    float*  Y12     = (float*)w;    w += align256((size_t)BATCH * M12 * TDIM * 4);
    float*  Y3      = (float*)w;    w += align256((size_t)BATCH * CIN * TDIM * 4);

    const dim3 blk32x8(32, 8, 1);

    // 1. resampling indices (batch-0 flf only)
    sampler_kernel<<<1, 256, 0, stream>>>(flf, idx);

    // 2. feature -> bf16 transposed (B, T, C)
    transpose_to_bf16_kernel<<<dim3(TDIM / 32, CIN / 32, BATCH), blk32x8, 0, stream>>>(
        feature, featT);

    // 3. weights -> bf16 (row-major layout already matches WMMA A frags)
    f32_to_bf16_kernel<<<512, 256, 0, stream>>>(W1, Wcat, (size_t)PDIM * CIN);
    f32_to_bf16_kernel<<<512, 256, 0, stream>>>(W2, Wcat + (size_t)PDIM * CIN,
                                                (size_t)2 * PDIM * CIN);
    f32_to_bf16_kernel<<<512, 256, 0, stream>>>(W3, W3b, (size_t)CIN * CMIX);
    copy_f32_kernel<<<(PDIM + 255) / 256, 256, 0, stream>>>(b1, biascat, PDIM);
    copy_f32_kernel<<<(2 * PDIM + 255) / 256, 256, 0, stream>>>(b2, biascat + PDIM,
                                                                2 * PDIM);

    // 4. sampled channels of mixedT
    gather_sampled_kernel<<<dim3(TDIM / 32, CIN / 32, BATCH), blk32x8, 0, stream>>>(
        flf, idx, mixedT);

    // 5. fused GEMM1+2 : Y12[b, 0..1535, t]
    gemm_bf16_wmma<<<dim3(TDIM / 128, M12 / 128, BATCH), 256, 0, stream>>>(
        Wcat, featT, biascat, Y12, M12, TDIM, CIN,
        (size_t)TDIM * CIN, (size_t)M12 * TDIM);

    // 6. group-norm stats: Y1 (rows 0..511, 16 ch/group), Y2 (rows 512..1535, 32 ch/group)
    gn_stats_kernel<<<dim3(NGROUPS, 1, BATCH), 256, 0, stream>>>(
        Y12, (size_t)M12 * TDIM, stats1, PDIM / NGROUPS);
    gn_stats_kernel<<<dim3(NGROUPS, 1, BATCH), 256, 0, stream>>>(
        Y12 + (size_t)PDIM * TDIM, (size_t)M12 * TDIM, stats2, 2 * PDIM / NGROUPS);

    // 7. apply: feat -> f32 output + bf16 mixedT[512..1535]; fm -> bf16 mixedT[1536..2047]
    gn_apply_tr_kernel<<<dim3(TDIM / 32, (2 * PDIM) / 32, BATCH), blk32x8, 0, stream>>>(
        Y12 + (size_t)PDIM * TDIM, (size_t)M12 * TDIM, stats2, g2, be2,
        out_feat, 2 * PDIM, mixedT, PDIM, 2 * PDIM / NGROUPS);
    gn_apply_tr_kernel<<<dim3(TDIM / 32, PDIM / 32, BATCH), blk32x8, 0, stream>>>(
        Y12, (size_t)M12 * TDIM, stats1, g1, be1,
        (float*)nullptr, PDIM, mixedT, PDIM + 2 * PDIM, PDIM / NGROUPS);

    // 8. GEMM3 : Y3[b, 0..511, t], K = 2048
    gemm_bf16_wmma<<<dim3(TDIM / 128, CIN / 128, BATCH), 256, 0, stream>>>(
        W3b, mixedT, b3, Y3, CIN, TDIM, CMIX,
        (size_t)TDIM * CMIX, (size_t)CIN * TDIM);

    // 9. final group-norm + relu -> mixed output
    gn_stats_kernel<<<dim3(NGROUPS, 1, BATCH), 256, 0, stream>>>(
        Y3, (size_t)CIN * TDIM, stats3, CIN / NGROUPS);
    gn_apply_kernel<<<dim3(TDIM / 256, CIN, BATCH), 256, 0, stream>>>(
        Y3, stats3, g3, be3, out_mixed, CIN, CIN / NGROUPS);

    (void)in_sizes; (void)n_in; (void)out_size; (void)ws_size;
}
